// LossUnsupervisedAngle_46119358825081
// MI455X (gfx1250) — matrix operand order repaired and checked by reading, not compiled, not run on hardware
//
#include <hip/hip_runtime.h>

typedef _Float16     v16h __attribute__((ext_vector_type(16)));
typedef _Float16     v4h  __attribute__((ext_vector_type(4)));
typedef float        v8f  __attribute__((ext_vector_type(8)));
typedef unsigned int v4u  __attribute__((ext_vector_type(4)));
typedef int          v4i  __attribute__((ext_vector_type(4)));

#define FEAT   512
#define NCLUST 1024
#define BM     128
#define BN     64
#define BROW   72    // padded b_lds row stride in halfs (64 + 8 pad)
#define KSTEPS 16    // FEAT / 32
#define NCHUNK 16    // NCLUST / BN
#define TOTAL_STEPS (KSTEPS * NCHUNK)

#if __has_builtin(__builtin_amdgcn_global_load_async_to_lds_b128) && \
    __has_builtin(__builtin_amdgcn_s_wait_asynccnt)
#define USE_ASYNC_LDS 1
#else
#define USE_ASYNC_LDS 0
#endif

union Frag16 { v16h h; v4u u[2]; };

__device__ inline float wave_reduce_add(float v) {
#pragma unroll
    for (int off = 16; off > 0; off >>= 1) v += __shfl_xor(v, off, 32);
    return v;
}

// 16-byte global -> LDS copy; async (ASYNCcnt-tracked) when available.
__device__ inline void copy16_to_lds(void* l, const void* g) {
#if USE_ASYNC_LDS
    __builtin_amdgcn_global_load_async_to_lds_b128(
        (__attribute__((address_space(1))) v4i*)(size_t)g,
        (__attribute__((address_space(3))) v4i*)l, 0, 0);
#else
    *(v4u*)l = *(const v4u*)g;
#endif
}
__device__ inline void wait_lds_copies() {
#if USE_ASYNC_LDS
    __builtin_amdgcn_s_wait_asynccnt(0);
#endif
}

// ---------------------------------------------------------------------------
// Kernel 1: normalize cluster centers m[c, :], write f16 TRANSPOSED mT[f][c]
// so each 64-cluster K row of a B slice is contiguous in memory. Also zero the global acc.
// ---------------------------------------------------------------------------
__global__ __launch_bounds__(256) void prep_m_kernel(const float* __restrict__ m,
                                                     _Float16* __restrict__ mT,
                                                     float* __restrict__ acc) {
    const int c = blockIdx.x;          // cluster index
    const int t = threadIdx.x;         // 0..255, covers 512 feats as 2 elems
    const int lane = t & 31, wid = t >> 5;

    float a = m[c * FEAT + t];
    float b = m[c * FEAT + 256 + t];
    float ss = wave_reduce_add(a * a + b * b);

    __shared__ float red[8];
    __shared__ float s_inv;
    if (lane == 0) red[wid] = ss;
    __syncthreads();
    if (t == 0) {
        float s = 0.f;
#pragma unroll
        for (int i = 0; i < 8; ++i) s += red[i];
        s_inv = 1.0f / fmaxf(sqrtf(s), 1e-12f);
        if (blockIdx.x == 0) acc[0] = 0.0f;   // stream-ordered before main kernel
    }
    __syncthreads();
    const float inv = s_inv;
    mT[(size_t)t         * NCLUST + c] = (_Float16)(a * inv);
    mT[(size_t)(t + 256) * NCLUST + c] = (_Float16)(b * inv);
}

// ---------------------------------------------------------------------------
// Kernel 2: fused  normalize(x) -> WMMA GEMM vs mT -> streaming softmax
// entropy (no max needed: logits are cosine sims in [-1,1]).
//   ent(row) = log(S) - T/S,  S = sum exp(y), T = sum y*exp(y)
// B staging is double-buffered with async global->LDS copies: one barrier
// per 32-wide K step, copy for step k+1 overlaps WMMA for step k.
// ---------------------------------------------------------------------------
__global__ __launch_bounds__(256) void angle_loss_kernel(const float* __restrict__ x,
                                                         const _Float16* __restrict__ mT,
                                                         float* __restrict__ acc) {
    extern __shared__ char smem[];
    _Float16* a_lds  = (_Float16*)smem;                              // [BM][FEAT] f16
    _Float16* b_lds0 = (_Float16*)(smem + BM * FEAT * 2);            // [32][BROW] f16
    _Float16* b_lds1 = b_lds0 + 32 * BROW;                           // buffer 1
    float*    s_lds  = (float*)(smem + BM * FEAT * 2 + 2 * 32 * BROW * 2);
    float*    t_lds  = s_lds + BM;
    float*    blk    = t_lds + BM;

    const int t      = threadIdx.x;
    const int lane   = t & 31;
    const int wid    = t >> 5;          // 0..7
    const int wave_m = wid & 3;         // 4-way split over M (32 rows each)
    const int wave_n = wid >> 2;        // 2-way split over N (32 cols each)
    const int row_base = blockIdx.x * BM;

    if (t < BM) { s_lds[t] = 0.f; t_lds[t] = 0.f; }
    if (t == 255) blk[0] = 0.f;

    // ---- Phase 1: coalesced load of 128 x-rows, L2-normalize, f16 into LDS
    // one float4 per lane == the first 128 floats of a row per wave iteration
#pragma unroll 1
    for (int rr = 0; rr < 16; ++rr) {
        const int r = wid * 16 + rr;
        const float4 v = ((const float4*)(x + (size_t)(row_base + r) * FEAT))[lane];
        float ss = wave_reduce_add(v.x * v.x + v.y * v.y + v.z * v.z + v.w * v.w);
        const float inv = 1.0f / fmaxf(sqrtf(ss), 1e-12f);
        v4h h;
        h[0] = (_Float16)(v.x * inv); h[1] = (_Float16)(v.y * inv);
        h[2] = (_Float16)(v.z * inv); h[3] = (_Float16)(v.w * inv);
        *(v4h*)(a_lds + (size_t)r * FEAT + lane * 4) = h;
    }

    // per-lane running entropy partials, indexed [m-subtile][acc VGPR row]
    float Sacc[2][8], Tacc[2][8];
#pragma unroll
    for (int mi = 0; mi < 2; ++mi)
#pragma unroll
        for (int i = 0; i < 8; ++i) { Sacc[mi][i] = 0.f; Tacc[mi][i] = 0.f; }

    // B staging assignment: 8 threads per 64-half K-row, 16 B each
    const int bk_row = t >> 3;          // 0..31 : K row of B slice
    const int bk_col = (t & 7) * 8;     // halfs
    const int bk_off = bk_row * BROW + bk_col;               // LDS halfs offset

    const v8f vzero = {0.f, 0.f, 0.f, 0.f, 0.f, 0.f, 0.f, 0.f};
    v8f accv[2][2];
#pragma unroll
    for (int mi = 0; mi < 2; ++mi)
#pragma unroll
        for (int ni = 0; ni < 2; ++ni) accv[mi][ni] = vzero;

    // global source of this thread's 16B of the B slice for step kk
    const _Float16* bsrc0 = mT + (size_t)bk_row * NCLUST + bk_col;
    // step kk: c0 = (kk/16)*BN, k0 = (kk%16)*32
    //   src = bsrc0 + k0*NCLUST + c0

    // prologue: fill buffer 0 with step 0
    copy16_to_lds(b_lds0 + bk_off, bsrc0);
    wait_lds_copies();
    __syncthreads();                     // also publishes a_lds

    const int half = lane >> 4;
    const int arow = wave_m * 32 + (lane & 15);
    int p = 0;

#pragma unroll 1
    for (int kk = 0; kk < TOTAL_STEPS; ++kk) {
        _Float16* bcur = p ? b_lds1 : b_lds0;
        _Float16* bnxt = p ? b_lds0 : b_lds1;

        // start async copy of step kk+1 into the spare buffer
        if (kk + 1 < TOTAL_STEPS) {
            const int kn = kk + 1;
            const _Float16* src =
                bsrc0 + (size_t)((kn & 15) * 32) * NCLUST + (kn >> 4) * BN;
            copy16_to_lds(bnxt + bk_off, src);
        }

        const int k0 = (kk & 15) * 32;

        // A frags: lane holds row M=lane%16; K halves {8h..8h+7, 16+8h..+7}
        Frag16 af[2];
#pragma unroll
        for (int mi = 0; mi < 2; ++mi) {
            const _Float16* ap =
                a_lds + (size_t)(arow + mi * 16) * FEAT + k0 + half * 8;
            af[mi].u[0] = *(const v4u*)(ap);
            af[mi].u[1] = *(const v4u*)(ap + 16);
        }
        // B frags: lane holds col N=lane%16; K halves {8h..8h+7, 16+8h..+7}
        Frag16 bf[2];
#pragma unroll
        for (int ni = 0; ni < 2; ++ni) {
            const _Float16* bp = bcur + wave_n * 32 + ni * 16 + (lane & 15);
#pragma unroll
            for (int s = 0; s < 8; ++s) {
                bf[ni].h[s]     = bp[(half * 8 + s) * BROW];
                bf[ni].h[8 + s] = bp[(16 + half * 8 + s) * BROW];
            }
        }
#pragma unroll
        for (int mi = 0; mi < 2; ++mi)
#pragma unroll
            for (int ni = 0; ni < 2; ++ni)
                accv[mi][ni] = __builtin_amdgcn_wmma_f32_16x16x32_f16(
                    false, af[mi].h, false, bf[ni].h,
                    (short)0, accv[mi][ni], false, false);

        // end of a 64-cluster chunk: fold logits into streaming S/T
        if ((kk & 15) == 15) {
#pragma unroll
            for (int mi = 0; mi < 2; ++mi)
#pragma unroll
                for (int ni = 0; ni < 2; ++ni)
#pragma unroll
                    for (int i = 0; i < 8; ++i) {
                        const float y = accv[mi][ni][i];
                        const float e = __expf(y);
                        Sacc[mi][i] += e;
                        Tacc[mi][i] += y * e;
                    }
#pragma unroll
            for (int mi = 0; mi < 2; ++mi)
#pragma unroll
                for (int ni = 0; ni < 2; ++ni) accv[mi][ni] = vzero;
        }

        // copies for kk+1 done + all waves done reading bcur
        wait_lds_copies();
        __syncthreads();
        p ^= 1;
    }

    // ---- reduce across the 16 N-lanes of each C-layout half, then LDS adds
#pragma unroll
    for (int mi = 0; mi < 2; ++mi)
#pragma unroll
        for (int i = 0; i < 8; ++i) {
            float s  = Sacc[mi][i];
            float tt = Tacc[mi][i];
#pragma unroll
            for (int m = 8; m > 0; m >>= 1) {
                s  += __shfl_xor(s,  m, 32);
                tt += __shfl_xor(tt, m, 32);
            }
            if ((lane & 15) == 0) {
                // C layout: lanes 0-15 -> row i, lanes 16-31 -> row i+8
                const int r = wave_m * 32 + mi * 16 + (lane >> 4) * 8 + i;
                atomicAdd(&s_lds[r], s);
                atomicAdd(&t_lds[r], tt);
            }
        }
    __syncthreads();

    if (t < BM) {
        const float S = s_lds[t];
        const float T = t_lds[t];
        const float ent = __logf(S) - T / S;
        atomicAdd(blk, ent);
    }
    __syncthreads();
    if (t == 0) atomicAdd(acc, blk[0]);
}

// ---------------------------------------------------------------------------
// Kernel 3: finalize (total, intra, inter)
// ---------------------------------------------------------------------------
__global__ void finalize_kernel(const float* __restrict__ acc,
                                float* __restrict__ out, float inv_n) {
    if (threadIdx.x == 0) {
        const float v = acc[0] * inv_n;
        out[0] = v;     // total
        out[1] = v;     // intra
        out[2] = 0.0f;  // inter
    }
}

extern "C" void kernel_launch(void* const* d_in, const int* in_sizes, int n_in,
                              void* d_out, int out_size, void* d_ws, size_t ws_size,
                              hipStream_t stream) {
    const float* x = (const float*)d_in[0];      // [N, 512] f32
    const float* m = (const float*)d_in[1];      // [1024, 512] f32
    float* out = (float*)d_out;                  // 3 floats

    float*    accp = (float*)d_ws;                         // 1 float accumulator
    _Float16* mT   = (_Float16*)((char*)d_ws + 256);       // [512][1024] f16

    const int N = in_sizes[0] / FEAT;            // 65536
    const int C = in_sizes[1] / FEAT;            // 1024

    prep_m_kernel<<<C, 256, 0, stream>>>(m, mT, accp);

    const size_t smem = (size_t)BM * FEAT * 2    // A tile f16
                      + 2 * 32 * BROW * 2        // B staging f16 (double buffer)
                      + BM * 4 * 2               // S/T per-row
                      + 16;                      // block accumulator
    angle_loss_kernel<<<N / BM, 256, smem, stream>>>(x, mT, accp);

    finalize_kernel<<<1, 32, 0, stream>>>(accp, out, 1.0f / (float)N);
}
